// Attention_86560770884077
// MI455X (gfx1250) — matrix operand order repaired and checked by reading, not compiled, not run on hardware
//
#include <hip/hip_runtime.h>
#include <stdint.h>
#include <utility>

// Problem constants (match reference)
constexpr int BB  = 2;
constexpr int TT  = 2048;
constexpr int CC  = 1024;
constexpr int HH  = 16;
constexpr int HS  = 64;          // head size
constexpr int N3  = 3 * CC;      // 3072

typedef __attribute__((ext_vector_type(16))) __bf16 v16bf;
typedef __attribute__((ext_vector_type(8)))  __bf16 v8bf;
typedef __attribute__((ext_vector_type(8)))  float  v8f;
typedef __attribute__((ext_vector_type(4)))  float  v4f;
typedef __attribute__((ext_vector_type(4)))  unsigned short v4u16;

union Frag16 { v16bf v; v8bf h[2]; };

#if __has_builtin(__builtin_amdgcn_global_load_async_to_lds_b128) && \
    __has_builtin(__builtin_amdgcn_s_wait_asynccnt)
#define HAVE_ASYNC_LDS 1
typedef int async_v4i __attribute__((vector_size(16)));
// global source: addrspace(1), non-const (builtin wants mutable int4*)
#define ASG(p) ((__attribute__((address_space(1))) async_v4i*)(unsigned short*)(uintptr_t)(p))
// LDS destination: addrspace(3) via direct pointer cast (proper addrspacecast)
#define ASL(p) ((__attribute__((address_space(3))) async_v4i*)(p))

// Emit N async b128 copies with template-constant immediate offsets I*STRIDE
// (applied by HW to BOTH the global and LDS addresses).
template <int STRIDE, int... I>
__device__ __forceinline__ void async_copy_impl(
    const unsigned short* g, unsigned short* l,
    std::integer_sequence<int, I...>) {
    (__builtin_amdgcn_global_load_async_to_lds_b128(
         ASG(g), ASL(l), I * STRIDE, 0), ...);
}
template <int N, int STRIDE>
__device__ __forceinline__ void async_copy(const unsigned short* g,
                                           unsigned short* l) {
    async_copy_impl<STRIDE>(g, l, std::make_integer_sequence<int, N>{});
}
#else
#define HAVE_ASYNC_LDS 0
#endif

__device__ __forceinline__ unsigned short f2bf(float f) {
    unsigned u = __float_as_uint(f);
    u += 0x7fffu + ((u >> 16) & 1u);     // round-to-nearest-even
    return (unsigned short)(u >> 16);
}

__device__ __forceinline__ v8f wmma_bf16(const Frag16& a, const Frag16& b, v8f c) {
    return __builtin_amdgcn_wmma_f32_16x16x32_bf16(
        false, a.v, false, b.v, (short)0, c, false, false);
}

// ---------------------------------------------------------------------------
// Stage 0a: fp32 -> bf16 elementwise convert (x), 4 elements/thread
// ---------------------------------------------------------------------------
__global__ void cvt_bf16_kernel(const float* __restrict__ src,
                                unsigned short* __restrict__ dst, int n4) {
    int i = blockIdx.x * blockDim.x + threadIdx.x;
    if (i < n4) {
        v4f f = *(const v4f*)(src + (size_t)i * 4);
        v4u16 o;
        #pragma unroll
        for (int k = 0; k < 4; ++k) o[k] = f2bf(f[k]);
        *(v4u16*)(dst + (size_t)i * 4) = o;
    }
}

// ---------------------------------------------------------------------------
// Stage 0b: fp32 [CC][COLS] -> bf16 transposed [COLS][CC].
// Each thread handles 4 rows x 1 col -> one contiguous 8B store.
// ---------------------------------------------------------------------------
template <int COLS>
__global__ void cvt_transpose_kernel(const float* __restrict__ src,
                                     unsigned short* __restrict__ dst) {
    static_assert(CC / 4 == 256, "shift below assumes CC==1024");
    int i = blockIdx.x * blockDim.x + threadIdx.x;    // [0, COLS*CC/4)
    if (i >= COLS * (CC / 4)) return;
    const int r4 = i & (CC / 4 - 1);
    const int c  = i >> 8;
    const int r  = r4 * 4;
    v4u16 o;
    #pragma unroll
    for (int k = 0; k < 4; ++k)
        o[k] = f2bf(src[(size_t)(r + k) * COLS + c]);
    *(v4u16*)(dst + (size_t)c * CC + r) = o;
}

// ---------------------------------------------------------------------------
// Stage 1: QKV GEMM.  One wave computes a 16x64 tile (4 accumulators).
// Per k-step: load A + all 4 B fragments (one load clause), then 4 WMMA.
// Scatters to Q[b,h,t,d] (scaled by 1/8), K[b,h,t,d], Vt[b,h,d,t].
// ---------------------------------------------------------------------------
__global__ __launch_bounds__(128) void qkv_gemm_kernel(
    const unsigned short* __restrict__ xb,     // [BB*TT][CC] bf16
    const unsigned short* __restrict__ WkqvT,  // [N3][CC] bf16 (transposed)
    unsigned short* __restrict__ Qo,           // [BB*HH][TT][HS]
    unsigned short* __restrict__ Ko,           // [BB*HH][TT][HS]
    unsigned short* __restrict__ Vto)          // [BB*HH][HS][TT]
{
    const int wave = threadIdx.x >> 5;
    const int lane = threadIdx.x & 31;
    const int half = lane >> 4;
    const int ln   = lane & 15;

    const int ng = blockIdx.x * 4 + wave;   // 0..47 (N in 64-wide groups)
    const int mt = blockIdx.y;              // 0..255
    const int m0 = mt * 16, n0 = ng * 64;

    const unsigned short* arow  = xb    + (size_t)(m0 + ln) * CC;
    const unsigned short* brow0 = WkqvT + (size_t)(n0 + ln) * CC;

    v8f acc[4] = {};
    for (int kk = 0; kk < CC; kk += 32) {
        Frag16 af, bfr[4];
        af.h[0] = *(const v8bf*)(arow + kk + 16 * half);
        af.h[1] = *(const v8bf*)(arow + kk + 16 * half + 8);
        #pragma unroll
        for (int j = 0; j < 4; ++j) {
            const unsigned short* br = brow0 + (size_t)(16 * j) * CC + kk + 16 * half;
            bfr[j].h[0] = *(const v8bf*)(br);
            bfr[j].h[1] = *(const v8bf*)(br + 8);
        }
        __builtin_prefetch(arow + kk + 1024, 0, 3);
        #pragma unroll
        for (int j = 0; j < 4; ++j)
            acc[j] = wmma_bf16(af, bfr[j], acc[j]);
    }

    // n0 is 64-aligned -> whole wave tile sits in one section & one head.
    const int sec = n0 >> 10;               // 0=q, 1=k, 2=v
    const int hh  = (n0 & (CC - 1)) >> 6;

    #pragma unroll
    for (int j = 0; j < 4; ++j) {
        const int d = j * 16 + ln;
        #pragma unroll
        for (int r = 0; r < 8; ++r) {
            const int g  = m0 + r + 8 * half;
            const int bI = g >> 11;
            const int t  = g & (TT - 1);
            const size_t bh = (size_t)bI * HH + hh;
            const float val = acc[j][r];
            if (sec == 0)      Qo [(bh * TT + t) * HS + d] = f2bf(val * 0.125f);
            else if (sec == 1) Ko [(bh * TT + t) * HS + d] = f2bf(val);
            else               Vto[(bh * HS + d) * TT + t] = f2bf(val);
        }
    }
}

// ---------------------------------------------------------------------------
// Stage 2: causal flash attention. One wave per (b,h, 16-query tile).
// 64-key blocks: S = Q·K^T (8 wmma), online softmax, O += P·V (8 wmma).
// K/V blocks are double-buffered in LDS via async-to-LDS; the ISA adds the
// instruction IOFFSET to BOTH the global and LDS addresses, so each lane
// computes one base per stream and the copies use template-imm offsets.
// ---------------------------------------------------------------------------
__global__ __launch_bounds__(128) void attn_kernel(
    const unsigned short* __restrict__ Q,   // [BB*HH][TT][HS] (pre-scaled)
    const unsigned short* __restrict__ K,   // [BB*HH][TT][HS]
    const unsigned short* __restrict__ Vt,  // [BB*HH][HS][TT]
    unsigned short* __restrict__ Obt)       // [BB*TT][CC] bf16, head-major cols
{
    extern __shared__ unsigned short smem[];

    const int wave = threadIdx.x >> 5;
    const int lane = threadIdx.x & 31;
    const int half = lane >> 4;
    const int ln   = lane & 15;

    const int bh = blockIdx.y;                    // 0..BB*HH-1
    const int bI = bh >> 4;
    const int hh = bh & (HH - 1);
    const int qt = blockIdx.x * 4 + wave;         // 0..127
    const int q0 = qt * 16;

    constexpr float LOG2E = 1.4426950408889634f;
    const float NEG_INF = -__builtin_inff();

#if HAVE_ASYNC_LDS
    // Per-wave LDS: K dbl-buf (2x8KB) + Vt dbl-buf (2x8KB) + P (2KB)
    unsigned short* wbase = smem + (size_t)wave * (4 * 4096 + 1024);
    unsigned short* kbuf[2] = { wbase,        wbase + 4096 };
    unsigned short* vbuf[2] = { wbase + 8192, wbase + 12288 };
    unsigned short* pl = wbase + 16384;

    // Per-lane global bases (k0-independent parts)
    const unsigned short* kg0 = K  + (size_t)bh * TT * HS + lane * 8;        // + k0*HS
    const unsigned short* vg0 = Vt + ((size_t)bh * HS + lane) * TT;          // + k0
    const unsigned short* vg1 = Vt + ((size_t)bh * HS + 32 + lane) * TT;     // + k0

    // Stage one 64-key block: K is a linear 8KB copy (lane-strided chunks,
    // imm offset +i*512 on both sides); V is 64 rows x 128B (lane = row,
    // imm offset +i*16 on both sides).
    auto stage = [&](int kb, int buf) {
        const int k0 = kb * 64;
        async_copy<16, 512>(kg0 + (size_t)k0 * HS, kbuf[buf] + lane * 8);
        async_copy<8, 16>(vg0 + k0, vbuf[buf] + lane * 64);
        async_copy<8, 16>(vg1 + k0, vbuf[buf] + (32 + lane) * 64);
    };
#else
    unsigned short* pl = smem + (size_t)wave * 1024;  // P staging only
#endif

    // Load Q fragments (16x64 -> two 16x32 A fragments), reused for all keys.
    Frag16 qa[2];
    {
        const unsigned short* row = Q + ((size_t)bh * TT + (q0 + ln)) * HS;
        #pragma unroll
        for (int f = 0; f < 2; ++f) {
            qa[f].h[0] = *(const v8bf*)(row + 32 * f + 16 * half);
            qa[f].h[1] = *(const v8bf*)(row + 32 * f + 16 * half + 8);
        }
    }

    v8f acc[4] = {};                 // O accumulator: 16 x 64 fp32
    float mrow[8], lrow[8];
    #pragma unroll
    for (int r = 0; r < 8; ++r) { mrow[r] = NEG_INF; lrow[r] = 0.f; }

    const int kbmax = (q0 + 15) >> 6;

#if HAVE_ASYNC_LDS
    stage(0, 0);
#endif

    for (int kbi = 0; kbi <= kbmax; ++kbi) {
        const int k0 = kbi * 64;

#if HAVE_ASYNC_LDS
        if (kbi < kbmax) {
            stage(kbi + 1, (kbi + 1) & 1);
            __builtin_amdgcn_s_wait_asynccnt(32);   // current buffer complete
        } else {
            __builtin_amdgcn_s_wait_asynccnt(0);
        }
        const unsigned short* kB = kbuf[kbi & 1];
        const unsigned short* vB = vbuf[kbi & 1];
#endif

        // --- S = Q K^T : four 16x16 tiles over the 64-key block ---
        v8f s[4];
        #pragma unroll
        for (int st = 0; st < 4; ++st) {
            v8f sa = {};
            #pragma unroll
            for (int ks = 0; ks < 2; ++ks) {
#if HAVE_ASYNC_LDS
                const unsigned short* krow =
                    kB + (st * 16 + ln) * 64 + 32 * ks + 16 * half;
#else
                const unsigned short* krow =
                    K + ((size_t)bh * TT + (k0 + st * 16 + ln)) * HS + 32 * ks + 16 * half;
#endif
                Frag16 kf;
                kf.h[0] = *(const v8bf*)(krow);
                kf.h[1] = *(const v8bf*)(krow + 8);
                sa = wmma_bf16(qa[ks], kf, sa);
            }
            s[st] = sa;
        }

        // --- causal mask + online softmax; P -> LDS as bf16 ---
        float alpha[8];
        #pragma unroll
        for (int r = 0; r < 8; ++r) {
            const int qrow = q0 + r + 8 * half;
            float mx = NEG_INF;
            #pragma unroll
            for (int st = 0; st < 4; ++st) {
                const int kcol = k0 + st * 16 + ln;
                float sv = s[st][r];
                if (kcol > qrow) sv = NEG_INF;
                s[st][r] = sv;
                mx = fmaxf(mx, sv);
            }
            #pragma unroll
            for (int off = 1; off < 16; off <<= 1)
                mx = fmaxf(mx, __shfl_xor(mx, off, 16));
            const float mnew = fmaxf(mrow[r], mx);
            alpha[r] = exp2f((mrow[r] - mnew) * LOG2E);
            float rsum = 0.f;
            #pragma unroll
            for (int st = 0; st < 4; ++st) {
                const float p = exp2f((s[st][r] - mnew) * LOG2E);
                rsum += p;
                pl[(r + 8 * half) * 64 + st * 16 + ln] = f2bf(p);
            }
            #pragma unroll
            for (int off = 1; off < 16; off <<= 1)
                rsum += __shfl_xor(rsum, off, 16);
            lrow[r] = lrow[r] * alpha[r] + rsum;
            mrow[r] = mnew;
        }

        // --- rescale O by alpha, then O += P · V (two k=32 steps) ---
        #pragma unroll
        for (int dt = 0; dt < 4; ++dt)
            #pragma unroll
            for (int r = 0; r < 8; ++r)
                acc[dt][r] *= alpha[r];

        #pragma unroll
        for (int ks = 0; ks < 2; ++ks) {
            Frag16 pf;
            const unsigned short* prow = pl + ln * 64 + 32 * ks;
            pf.h[0] = *(const v8bf*)(prow + 16 * half);
            pf.h[1] = *(const v8bf*)(prow + 16 * half + 8);
            #pragma unroll
            for (int dt = 0; dt < 4; ++dt) {
#if HAVE_ASYNC_LDS
                const unsigned short* vrow =
                    vB + (dt * 16 + ln) * 64 + 32 * ks + 16 * half;
#else
                const unsigned short* vrow =
                    Vt + ((size_t)bh * HS + dt * 16 + ln) * TT + k0 + 32 * ks + 16 * half;
#endif
                Frag16 vf;
                vf.h[0] = *(const v8bf*)(vrow);
                vf.h[1] = *(const v8bf*)(vrow + 8);
                acc[dt] = wmma_bf16(pf, vf, acc[dt]);
            }
        }
    }

    // --- epilogue: O *= 1/l, store bf16 into (B*T, C) head-major layout ---
    #pragma unroll
    for (int r = 0; r < 8; ++r) {
        const float inv = 1.0f / lrow[r];
        const int t = q0 + r + 8 * half;
        #pragma unroll
        for (int dt = 0; dt < 4; ++dt) {
            const float o = acc[dt][r] * inv;
            Obt[((size_t)(bI * TT + t)) * CC + hh * HS + dt * 16 + ln] = f2bf(o);
        }
    }
}

// ---------------------------------------------------------------------------
// Stage 3: output projection  y = O @ W_proj + b_proj  (fp32 out), 16x64 tiles
// ---------------------------------------------------------------------------
__global__ __launch_bounds__(128) void proj_gemm_kernel(
    const unsigned short* __restrict__ Obt,    // [BB*TT][CC] bf16
    const unsigned short* __restrict__ WprojT, // [CC][CC] bf16 (transposed)
    const float* __restrict__ bias,            // [CC]
    float* __restrict__ out)                   // [BB*TT][CC] fp32
{
    const int wave = threadIdx.x >> 5;
    const int lane = threadIdx.x & 31;
    const int half = lane >> 4;
    const int ln   = lane & 15;

    const int ng = blockIdx.x * 4 + wave;   // 0..15 (N in 64-wide groups)
    const int mt = blockIdx.y;              // 0..255
    const int m0 = mt * 16, n0 = ng * 64;

    const unsigned short* arow  = Obt    + (size_t)(m0 + ln) * CC;
    const unsigned short* brow0 = WprojT + (size_t)(n0 + ln) * CC;

    v8f acc[4] = {};
    for (int kk = 0; kk < CC; kk += 32) {
        Frag16 af, bfr[4];
        af.h[0] = *(const v8bf*)(arow + kk + 16 * half);
        af.h[1] = *(const v8bf*)(arow + kk + 16 * half + 8);
        #pragma unroll
        for (int j = 0; j < 4; ++j) {
            const unsigned short* br = brow0 + (size_t)(16 * j) * CC + kk + 16 * half;
            bfr[j].h[0] = *(const v8bf*)(br);
            bfr[j].h[1] = *(const v8bf*)(br + 8);
        }
        __builtin_prefetch(arow + kk + 1024, 0, 3);
        #pragma unroll
        for (int j = 0; j < 4; ++j)
            acc[j] = wmma_bf16(af, bfr[j], acc[j]);
    }

    #pragma unroll
    for (int j = 0; j < 4; ++j) {
        const float bn = bias[n0 + j * 16 + ln];
        #pragma unroll
        for (int r = 0; r < 8; ++r) {
            const int g = m0 + r + 8 * half;
            out[(size_t)g * CC + n0 + j * 16 + ln] = acc[j][r] + bn;
        }
    }
}

// ---------------------------------------------------------------------------
// Launcher
// ---------------------------------------------------------------------------
extern "C" void kernel_launch(void* const* d_in, const int* in_sizes, int n_in,
                              void* d_out, int out_size, void* d_ws, size_t ws_size,
                              hipStream_t stream) {
    const float* x      = (const float*)d_in[0];   // (B,T,C)
    const float* W_kqv  = (const float*)d_in[1];   // (C,3C)
    const float* W_proj = (const float*)d_in[2];   // (C,C)
    const float* b_proj = (const float*)d_in[3];   // (C,)
    float* out = (float*)d_out;

    // Workspace carve-up (bf16 buffers)
    uint8_t* w = (uint8_t*)d_ws;
    unsigned short* xb     = (unsigned short*)w; w += (size_t)BB * TT * CC * 2;
    unsigned short* WkqvT  = (unsigned short*)w; w += (size_t)N3 * CC * 2;
    unsigned short* WprojT = (unsigned short*)w; w += (size_t)CC * CC * 2;
    unsigned short* Qb     = (unsigned short*)w; w += (size_t)BB * HH * TT * HS * 2;
    unsigned short* Kb     = (unsigned short*)w; w += (size_t)BB * HH * TT * HS * 2;
    unsigned short* Vtb    = (unsigned short*)w; w += (size_t)BB * HH * HS * TT * 2;
    unsigned short* Obt    = (unsigned short*)w; w += (size_t)BB * TT * CC * 2;

    // Stage 0: conversions
    {
        int n4 = (BB * TT * CC) / 4;
        cvt_bf16_kernel<<<(n4 + 255) / 256, 256, 0, stream>>>(x, xb, n4);
    }
    {
        int n = N3 * (CC / 4);
        cvt_transpose_kernel<N3><<<(n + 255) / 256, 256, 0, stream>>>(W_kqv, WkqvT);
    }
    {
        int n = CC * (CC / 4);
        cvt_transpose_kernel<CC><<<(n + 255) / 256, 256, 0, stream>>>(W_proj, WprojT);
    }

    // Stage 1: QKV GEMM  (N3/64 = 48 wave-groups -> grid.x 12; M tiles 256)
    qkv_gemm_kernel<<<dim3(N3 / 64 / 4, (BB * TT) / 16), 128, 0, stream>>>(
        xb, WkqvT, Qb, Kb, Vtb);

    // Stage 2: flash attention (q-tiles 128 -> grid.x 32; bh 32)
#if HAVE_ASYNC_LDS
    size_t attnSmem = (size_t)4 * (4 * 4096 + 1024) * sizeof(unsigned short); // 136KB
#else
    size_t attnSmem = (size_t)4 * 1024 * sizeof(unsigned short);              // 8KB
#endif
    attn_kernel<<<dim3((TT / 16) / 4, BB * HH), 128, attnSmem, stream>>>(
        Qb, Kb, Vtb, Obt);

    // Stage 3: projection + bias (CC/64 = 16 wave-groups -> grid.x 4)
    proj_gemm_kernel<<<dim3(CC / 64 / 4, (BB * TT) / 16), 128, 0, stream>>>(
        Obt, WprojT, b_proj, out);
}
